// contextEncoder_64269890617925
// MI455X (gfx1250) — compile-verified, matrix operand order checked by reading, not run on hardware
//
#include <hip/hip_runtime.h>
#include <hip/hip_bf16.h>
#include <math.h>

// ---------------------------------------------------------------------------
// BERT-base encoder forward for MI455X (gfx1250, wave32, WMMA).
// All GEMMs run on v_wmma_f32_16x16x32_bf16 (fp32 accumulate).
// Round-2 changes: templated N/K so all tile offsets fold into IOFFSET
// immediates; per-lane base pointers hoisted (incl. kh*8); LDS-tiled
// coalesced weight transpose.
// ---------------------------------------------------------------------------

typedef __bf16 bf16;
typedef bf16  bf16x2  __attribute__((ext_vector_type(2)));
typedef bf16  bf16x8  __attribute__((ext_vector_type(8)));
typedef bf16  bf16x16 __attribute__((ext_vector_type(16)));
typedef float f32x8   __attribute__((ext_vector_type(8)));

#define B_   32
#define S_   256
#define W_   128
#define H_   768
#define L_   12
#define NH_  12
#define DH_  64
#define FF_  3072
#define TOK_ (B_ * S_)   // 8192 token rows

// ---- fragment load (ISA 7.12.2, 16-bit A/B 16x32 layout) -------------------
// Caller passes a pointer that already includes lane_row*stride + kh*8 so the
// two 16-byte loads use pure immediate offsets.
__device__ __forceinline__ bf16x16 frag_ld(const bf16* p) {
  bf16x8 lo = *(const bf16x8*)(p);
  bf16x8 hi = *(const bf16x8*)(p + 16);
  return __builtin_shufflevector(lo, hi, 0, 1, 2, 3, 4, 5, 6, 7,
                                 8, 9, 10, 11, 12, 13, 14, 15);
}

__device__ __forceinline__ f32x8 wmma_bf16(bf16x16 a, bf16x16 b, f32x8 c) {
  return __builtin_amdgcn_wmma_f32_16x16x32_bf16(
      /*neg_a=*/false, a, /*neg_b=*/false, b,
      /*c_mod=*/(short)0, c, /*reuse_a=*/false, /*reuse_b=*/false);
}

__device__ __forceinline__ float gelu_exact(float x) {
  return 0.5f * x * (1.0f + erff(x * 0.70710678118654752f));
}

// half-wave (16 lane) reductions: xor masks 1,2,4,8 keep bit4 -> stay in half.
__device__ __forceinline__ float half_max16(float v) {
  v = fmaxf(v, __shfl_xor(v, 1, 32));
  v = fmaxf(v, __shfl_xor(v, 2, 32));
  v = fmaxf(v, __shfl_xor(v, 4, 32));
  v = fmaxf(v, __shfl_xor(v, 8, 32));
  return v;
}
__device__ __forceinline__ float half_sum16(float v) {
  v += __shfl_xor(v, 1, 32);
  v += __shfl_xor(v, 2, 32);
  v += __shfl_xor(v, 4, 32);
  v += __shfl_xor(v, 8, 32);
  return v;
}

__device__ __forceinline__ float block_sum256(float v, float* red) {
  int tid = threadIdx.x;
  red[tid] = v;
  __syncthreads();
#pragma unroll
  for (int s = 128; s > 0; s >>= 1) {
    if (tid < s) red[tid] += red[tid + s];
    __syncthreads();
  }
  float r = red[0];
  __syncthreads();
  return r;
}

// ---------------------------------------------------------------------------
// Coalesced weight convert+transpose via LDS: fp32 in [K,N] -> bf16 out [N,K].
// 64(K) x 32(N) tile: 128B coalesced fp32 reads, packed bf16x2 coalesced
// writes (128B per wave).  Avoids the 32x read amplification of a naive
// strided transpose (which would cost ~0.5 ms/pass at 23.3 TB/s).
// ---------------------------------------------------------------------------
template <int K, int N>
__global__ __launch_bounds__(256) void transpose_conv_kernel(
    const float* __restrict__ in, bf16* __restrict__ out) {
  __shared__ bf16 tile[64][33];
  const int tx = threadIdx.x & 31;
  const int ty = threadIdx.x >> 5;
  const int ntn = N / 32;
  const int k0 = (blockIdx.x / ntn) * 64;
  const int n0 = (blockIdx.x % ntn) * 32;
#pragma unroll
  for (int i = 0; i < 8; i++) {
    const int k = ty + i * 8;
    tile[k][tx] = (bf16)in[(size_t)(k0 + k) * N + (n0 + tx)];
  }
  __syncthreads();
#pragma unroll
  for (int i = 0; i < 4; i++) {
    const int n = ty + i * 8;
    bf16x2 v;
    v[0] = tile[2 * tx + 0][n];
    v[1] = tile[2 * tx + 1][n];
    *(bf16x2*)(out + (size_t)(n0 + n) * K + k0 + 2 * tx) = v;
  }
}

// ---------------------------------------------------------------------------
// GEMM: C[M,N] = A[M,K](bf16 RM) * Bt[N,K](bf16 RM)^T + bias[N]
// 8 waves (4 along M x 2 along N); wave tile 32x64 (2x4 accumulators);
// block tile 128x128.  N,K are template constants -> all intra-tile offsets
// are IOFFSET immediates; only one induction pointer pair per operand.
// MODE 0: bf16 row-major (optional GELU)  MODE 1: fp32 row-major
// MODE 2: bf16 per-head transposed -> Vt[B,NH,DH,S] (packed b128 stores)
// ---------------------------------------------------------------------------
template <int MODE, int GELU, int N, int K>
__global__ __launch_bounds__(256) void gemm_bf16_kernel(
    const bf16* __restrict__ A, const bf16* __restrict__ Bt,
    const float* __restrict__ bias, float* __restrict__ outF,
    bf16* __restrict__ outB) {
  const int lane = threadIdx.x & 31;
  const int wave = threadIdx.x >> 5;
  const int wm = wave & 3;
  const int wn = wave >> 2;
  const int nBlocks = N >> 7;
  const int bm = blockIdx.x / nBlocks;
  const int bn = blockIdx.x % nBlocks;
  const int m0 = bm * 128 + wm * 32;
  const int n0 = bn * 128 + wn * 64;
  const int kh = lane >> 4;
  const int lr = lane & 15;

  f32x8 acc[2][4] = {};

  // per-lane base pointers (include lr*K + kh*8); everything else immediate
  const bf16* pa = A + (size_t)(m0 + lr) * K + kh * 8;
  const bf16* pb = Bt + (size_t)(n0 + lr) * K + kh * 8;

  for (int k0 = 0; k0 < K; k0 += 32) {
    bf16x16 a0 = frag_ld(pa + k0);
    bf16x16 a1 = frag_ld(pa + 16 * K + k0);
    bf16x16 b0 = frag_ld(pb + k0);
    bf16x16 b1 = frag_ld(pb + 16 * K + k0);
    bf16x16 b2 = frag_ld(pb + 32 * K + k0);
    bf16x16 b3 = frag_ld(pb + 48 * K + k0);
    acc[0][0] = wmma_bf16(a0, b0, acc[0][0]);
    acc[1][0] = wmma_bf16(a1, b0, acc[1][0]);
    acc[0][1] = wmma_bf16(a0, b1, acc[0][1]);
    acc[1][1] = wmma_bf16(a1, b1, acc[1][1]);
    acc[0][2] = wmma_bf16(a0, b2, acc[0][2]);
    acc[1][2] = wmma_bf16(a1, b2, acc[1][2]);
    acc[0][3] = wmma_bf16(a0, b3, acc[0][3]);
    acc[1][3] = wmma_bf16(a1, b3, acc[1][3]);
  }

  // bias: one base, immediate per-j offsets
  float bv[4];
#pragma unroll
  for (int j = 0; j < 4; j++) bv[j] = bias[n0 + 16 * j + lr];

  // C layout (ISA 7.12.2): VGPR r, lane half -> M = r + 8*kh, N = lr.
  if (MODE == 2) {
    // head = n0>>6 and batch = m0>>8 are wave-constant (tiles never cross).
    bf16* vb = outB +
               ((size_t)((m0 >> 8) * NH_ + (n0 >> 6)) * DH_ + lr) * S_ +
               (m0 & 255) + 8 * kh;
#pragma unroll
    for (int i = 0; i < 2; i++) {
#pragma unroll
      for (int j = 0; j < 4; j++) {
        bf16x8 pk;
#pragma unroll
        for (int r = 0; r < 8; r++) pk[r] = (bf16)(acc[i][j][r] + bv[j]);
        *(bf16x8*)(vb + (size_t)(16 * j) * S_ + 16 * i) = pk;
      }
    }
  } else {
    const size_t base = (size_t)(m0 + 8 * kh) * N + n0 + lr;
#pragma unroll
    for (int i = 0; i < 2; i++) {
#pragma unroll
      for (int j = 0; j < 4; j++) {
#pragma unroll
        for (int r = 0; r < 8; r++) {
          float v = acc[i][j][r] + bv[j];
          if (GELU) v = gelu_exact(v);
          const size_t o = base + (size_t)(16 * i + r) * N + 16 * j;
          if (MODE == 1)
            outF[o] = v;
          else
            outB[o] = (bf16)v;
        }
      }
    }
  }
}

// ---------------------------------------------------------------------------
// Embedding gather + LayerNorm -> xf (fp32 residual stream) + xb (bf16 GEMM A)
// ---------------------------------------------------------------------------
__global__ __launch_bounds__(256) void embed_ln_kernel(
    const float* __restrict__ word_emb, const float* __restrict__ pos_emb,
    const float* __restrict__ type_emb, const int* __restrict__ sents,
    const float* __restrict__ g, const float* __restrict__ bta,
    float* __restrict__ xf, bf16* __restrict__ xb) {
  __shared__ float red[256];
  const int tok = blockIdx.x;
  const int tid = threadIdx.x;
  const int s = tok & (S_ - 1);
  const int id = sents[tok];
  float t[3];
#pragma unroll
  for (int i = 0; i < 3; i++) {
    int c = tid + i * 256;
    t[i] = word_emb[(size_t)id * H_ + c] + pos_emb[(size_t)s * H_ + c] +
           type_emb[c];
  }
  float mean = block_sum256(t[0] + t[1] + t[2], red) * (1.0f / H_);
  float vs = 0.f;
#pragma unroll
  for (int i = 0; i < 3; i++) {
    float d = t[i] - mean;
    vs += d * d;
  }
  float var = block_sum256(vs, red) * (1.0f / H_);
  float inv = rsqrtf(var + 1e-12f);
#pragma unroll
  for (int i = 0; i < 3; i++) {
    int c = tid + i * 256;
    float o = (t[i] - mean) * inv * g[c] + bta[c];
    xf[(size_t)tok * H_ + c] = o;
    xb[(size_t)tok * H_ + c] = (bf16)o;
  }
}

// x = LN(x + y) (y already holds GEMM-out + bias); updates xf and xb.
__global__ __launch_bounds__(256) void ln_residual_kernel(
    float* __restrict__ xf, const float* __restrict__ yf,
    const float* __restrict__ g, const float* __restrict__ bta,
    bf16* __restrict__ xb) {
  __shared__ float red[256];
  const size_t row = blockIdx.x;
  const int tid = threadIdx.x;
  float t[3];
#pragma unroll
  for (int i = 0; i < 3; i++) {
    int c = tid + i * 256;
    t[i] = xf[row * H_ + c] + yf[row * H_ + c];
  }
  float mean = block_sum256(t[0] + t[1] + t[2], red) * (1.0f / H_);
  float vs = 0.f;
#pragma unroll
  for (int i = 0; i < 3; i++) {
    float d = t[i] - mean;
    vs += d * d;
  }
  float var = block_sum256(vs, red) * (1.0f / H_);
  float inv = rsqrtf(var + 1e-12f);
#pragma unroll
  for (int i = 0; i < 3; i++) {
    int c = tid + i * 256;
    float o = (t[i] - mean) * inv * g[c] + bta[c];
    xf[row * H_ + c] = o;
    xb[row * H_ + c] = (bf16)o;
  }
}

// ---------------------------------------------------------------------------
// Fused attention for one (batch, head, 128-query block).
// 8 waves; wave w owns 16 queries.  scores = QK^T via WMMA, in-register
// softmax (half-wave shuffles), probs staged bf16 in LDS (C-layout b16 stores
// -> A-layout b128 loads, same-wave DS ordering => no barrier), ctx = P*V via
// WMMA with Vt[B,NH,DH,S] B-fragments.  All tile offsets are immediates off
// per-lane base pointers.
// ---------------------------------------------------------------------------
__global__ __launch_bounds__(256) void attention_kernel(
    const bf16* __restrict__ qb, const bf16* __restrict__ kb,
    const bf16* __restrict__ vtb, const int* __restrict__ mask,
    bf16* __restrict__ ctxb) {
  extern __shared__ char smem_raw[];
  bf16* probs = (bf16*)smem_raw;  // 128 rows x 256 keys, bf16 (64 KB)

  const int lane = threadIdx.x & 31;
  const int wave = threadIdx.x >> 5;
  const int b = blockIdx.z;
  const int h = blockIdx.y;
  const int q0 = blockIdx.x * 128 + wave * 16;
  const int kh = lane >> 4;
  const int lr = lane & 15;

  // ---- scores: 16 queries x 256 keys --------------------------------------
  const bf16* qp = qb + (size_t)(b * S_ + q0 + lr) * H_ + h * DH_ + 8 * kh;
  bf16x16 qa0 = frag_ld(qp);
  bf16x16 qa1 = frag_ld(qp + 32);

  const bf16* kp = kb + (size_t)(b * S_ + lr) * H_ + h * DH_ + 8 * kh;
  const int* mp = mask + b * S_ + lr;

  f32x8 sc[16];
#pragma unroll
  for (int t = 0; t < 16; t++) {
    bf16x16 kb0 = frag_ld(kp + (size_t)(t * 16) * H_);
    bf16x16 kb1 = frag_ld(kp + (size_t)(t * 16) * H_ + 32);
    f32x8 a = {};
    a = wmma_bf16(qa0, kb0, a);
    a = wmma_bf16(qa1, kb1, a);
    const float mk = mp[t * 16] ? 0.f : -1e9f;
#pragma unroll
    for (int r = 0; r < 8; r++) sc[t][r] = a[r] * 0.125f + mk;  // 1/sqrt(64)
  }

  // ---- softmax over 256 keys per query row --------------------------------
  bf16* prow_st = probs + (size_t)(wave * 16 + 8 * kh) * 256 + lr;
#pragma unroll
  for (int r = 0; r < 8; r++) {
    float mx = -3.4e38f;
#pragma unroll
    for (int t = 0; t < 16; t++) mx = fmaxf(mx, sc[t][r]);
    mx = half_max16(mx);
    float sum = 0.f;
#pragma unroll
    for (int t = 0; t < 16; t++) {
      float e = __expf(sc[t][r] - mx);
      sc[t][r] = e;
      sum += e;
    }
    sum = half_sum16(sum);
    const float invs = 1.0f / sum;
#pragma unroll
    for (int t = 0; t < 16; t++)
      prow_st[r * 256 + t * 16] = (bf16)(sc[t][r] * invs);
  }

  // ---- ctx = probs(16x256) * V(256x64) ------------------------------------
  f32x8 cacc[4] = {};
  const bf16* pp = probs + (size_t)(wave * 16 + lr) * 256 + 8 * kh;
  const bf16* vp = vtb + ((size_t)(b * NH_ + h) * DH_ + lr) * S_ + 8 * kh;
#pragma unroll
  for (int kk = 0; kk < 8; kk++) {
    bf16x16 pa = frag_ld(pp + kk * 32);
#pragma unroll
    for (int j = 0; j < 4; j++) {
      bf16x16 vbf = frag_ld(vp + (size_t)(j * 16) * S_ + kk * 32);
      cacc[j] = wmma_bf16(pa, vbf, cacc[j]);
    }
  }
  bf16* cp = ctxb + (size_t)(b * S_ + q0 + 8 * kh) * H_ + h * DH_ + lr;
#pragma unroll
  for (int j = 0; j < 4; j++)
#pragma unroll
    for (int r = 0; r < 8; r++)
      cp[(size_t)r * H_ + j * 16] = (bf16)cacc[j][r];
}

// ---------------------------------------------------------------------------
// Segment mean: seg_ids are sorted per batch -> binary search [lo,hi) for
// word w; deterministic, no atomics.
// ---------------------------------------------------------------------------
__global__ __launch_bounds__(256) void seg_mean_kernel(
    const float* __restrict__ xf, const int* __restrict__ seg,
    float* __restrict__ out) {
  const int b = blockIdx.x / W_;
  const int w = blockIdx.x % W_;
  const int* sb = seg + b * S_;
  int lo, hi;
  {
    int l = 0, r = S_;
    while (l < r) { int m = (l + r) >> 1; if (sb[m] < w) l = m + 1; else r = m; }
    lo = l;
  }
  {
    int l = 0, r = S_;
    while (l < r) { int m = (l + r) >> 1; if (sb[m] <= w) l = m + 1; else r = m; }
    hi = l;
  }
  const int cnt = hi - lo;
  const float invc = 1.0f / (float)(cnt > 0 ? cnt : 1);
  for (int c = threadIdx.x; c < H_; c += 256) {
    float s = 0.f;
    for (int i = lo; i < hi; i++) s += xf[(size_t)(b * S_ + i) * H_ + c];
    out[(size_t)(b * W_ + w) * H_ + c] = s * invc;
  }
}

// ---------------------------------------------------------------------------
// Host driver
// ---------------------------------------------------------------------------
extern "C" void kernel_launch(void* const* d_in, const int* in_sizes, int n_in,
                              void* d_out, int out_size, void* d_ws,
                              size_t ws_size, hipStream_t stream) {
  // setup_inputs() insertion order (params dict recursed in insertion order):
  const float* word_emb = (const float*)d_in[0];
  const float* pos_emb  = (const float*)d_in[1];
  const float* type_emb = (const float*)d_in[2];
  const float* eg       = (const float*)d_in[3];
  const float* ebias    = (const float*)d_in[4];
  const float* Wq = (const float*)d_in[5];  const float* bq = (const float*)d_in[6];
  const float* Wk = (const float*)d_in[7];  const float* bk = (const float*)d_in[8];
  const float* Wv = (const float*)d_in[9];  const float* bv = (const float*)d_in[10];
  const float* Wo = (const float*)d_in[11]; const float* bo = (const float*)d_in[12];
  const float* ln1g = (const float*)d_in[13]; const float* ln1b = (const float*)d_in[14];
  const float* W1 = (const float*)d_in[15]; const float* b1 = (const float*)d_in[16];
  const float* W2 = (const float*)d_in[17]; const float* b2 = (const float*)d_in[18];
  const float* ln2g = (const float*)d_in[19]; const float* ln2b = (const float*)d_in[20];
  const int* sents = (const int*)d_in[21];  // harness delivers integers as int32
  const int* mask  = (const int*)d_in[22];
  const int* seg   = (const int*)d_in[23];
  float* out = (float*)d_out;

  char* ws = (char*)d_ws;
  size_t off = 0;
  auto alloc = [&](size_t bytes) -> void* {
    void* p = ws + off;
    off += (bytes + 255) & ~(size_t)255;
    return p;
  };
  float* xf   = (float*)alloc((size_t)TOK_ * H_ * 4);   // fp32 residual stream
  float* yf   = (float*)alloc((size_t)TOK_ * H_ * 4);   // fp32 GEMM out (+bias)
  bf16* xb    = (bf16*)alloc((size_t)TOK_ * H_ * 2);    // bf16 copy of x
  bf16* qbuf  = (bf16*)alloc((size_t)TOK_ * H_ * 2);
  bf16* kbuf  = (bf16*)alloc((size_t)TOK_ * H_ * 2);
  bf16* vtb   = (bf16*)alloc((size_t)TOK_ * H_ * 2);    // V^T per head [B,NH,DH,S]
  bf16* ctxb  = (bf16*)alloc((size_t)TOK_ * H_ * 2);
  bf16* ffb   = (bf16*)alloc((size_t)TOK_ * FF_ * 2);
  bf16* wt    = (bf16*)alloc((size_t)H_ * FF_ * 2);     // staged bf16 Wt[N,K]

  embed_ln_kernel<<<TOK_, 256, 0, stream>>>(word_emb, pos_emb, type_emb, sents,
                                            eg, ebias, xf, xb);

  const int gProj = (TOK_ / 128) * (H_ / 128);          // 384
  const int gFF1  = (TOK_ / 128) * (FF_ / 128);         // 1536
  const int gTrHH = (H_ / 64) * (H_ / 32);              // 288
  const int gTrHF = (H_ / 64) * (FF_ / 32);             // 1152
  const int gTrFH = (FF_ / 64) * (H_ / 32);             // 1152

  for (int l = 0; l < L_; l++) {
    const size_t oHH = (size_t)l * H_ * H_;
    const size_t oHF = (size_t)l * H_ * FF_;

    // Q
    transpose_conv_kernel<H_, H_><<<gTrHH, 256, 0, stream>>>(Wq + oHH, wt);
    gemm_bf16_kernel<0, 0, H_, H_><<<gProj, 256, 0, stream>>>(
        xb, wt, bq + (size_t)l * H_, nullptr, qbuf);
    // K
    transpose_conv_kernel<H_, H_><<<gTrHH, 256, 0, stream>>>(Wk + oHH, wt);
    gemm_bf16_kernel<0, 0, H_, H_><<<gProj, 256, 0, stream>>>(
        xb, wt, bk + (size_t)l * H_, nullptr, kbuf);
    // V -> transposed-per-head layout for contiguous PV B-fragments
    transpose_conv_kernel<H_, H_><<<gTrHH, 256, 0, stream>>>(Wv + oHH, wt);
    gemm_bf16_kernel<2, 0, H_, H_><<<gProj, 256, 0, stream>>>(
        xb, wt, bv + (size_t)l * H_, nullptr, vtb);

    // fused attention (64 KB dynamic LDS for probs)
    attention_kernel<<<dim3(S_ / 128, NH_, B_), 256, 64 * 1024, stream>>>(
        qbuf, kbuf, vtb, mask, ctxb);

    // O projection -> yf (fp32, bias added), then LN1 residual
    transpose_conv_kernel<H_, H_><<<gTrHH, 256, 0, stream>>>(Wo + oHH, wt);
    gemm_bf16_kernel<1, 0, H_, H_><<<gProj, 256, 0, stream>>>(
        ctxb, wt, bo + (size_t)l * H_, yf, nullptr);
    ln_residual_kernel<<<TOK_, 256, 0, stream>>>(
        xf, yf, ln1g + (size_t)l * H_, ln1b + (size_t)l * H_, xb);

    // FFN1 (H->FF) + exact GELU, bf16 out
    transpose_conv_kernel<H_, FF_><<<gTrHF, 256, 0, stream>>>(W1 + oHF, wt);
    gemm_bf16_kernel<0, 1, FF_, H_><<<gFF1, 256, 0, stream>>>(
        xb, wt, b1 + (size_t)l * FF_, nullptr, ffb);

    // FFN2 (FF->H) -> yf, then LN2 residual
    transpose_conv_kernel<FF_, H_><<<gTrFH, 256, 0, stream>>>(W2 + oHF, wt);
    gemm_bf16_kernel<1, 0, H_, FF_><<<gProj, 256, 0, stream>>>(
        ffb, wt, b2 + (size_t)l * H_, yf, nullptr);
    ln_residual_kernel<<<TOK_, 256, 0, stream>>>(
        xf, yf, ln2g + (size_t)l * H_, ln2b + (size_t)l * H_, xb);
  }

  // scatter-mean of wordpieces into word slots (seg_ids sorted per batch)
  seg_mean_kernel<<<B_ * W_, 256, 0, stream>>>(xf, seg, out);
}
